// AttnMessageAggregator_57844619542973
// MI455X (gfx1250) — matrix operand order repaired and checked by reading, not compiled, not run on hardware
//
#include <hip/hip_runtime.h>

// ---------------------------------------------------------------------------
// AttnMessageAggregator for MI455X (gfx1250, wave32, WMMA + async-LDS)
//   Q   = (memory @ Wq^T + bq) * 0.125          -> batched WMMA GEMM (pre)
//   K/V = msg @ {Wk,Wv}^T + b   per node        -> fused WMMA in attn kernel
//   ctx = softmax(q.K^T, masked) @ V            -> fused (VALU, small)
//   agg = ctx @ Wo^T + bo                       -> batched WMMA GEMM (post)
//   out = agg @ Wm^T + bm                       -> batched WMMA GEMM (post)
// ---------------------------------------------------------------------------

#define N_NODES 4096
#define SEQ_L   128
#define DMODEL  512
#define NHEAD   8
#define HDIM    64
#define SMS     520   // LDS row stride in bf16 elems (bank-conflict-free, 16B aligned)

typedef __attribute__((ext_vector_type(16))) __bf16 bf16x16;
typedef __attribute__((ext_vector_type(8)))  __bf16 bf16x8;
typedef __attribute__((ext_vector_type(8)))  float  f32x8;

union AFrag { bf16x16 v; bf16x8 h[2]; };

__device__ __forceinline__ unsigned short f2bf_u16(float f) {
  unsigned u = __builtin_bit_cast(unsigned, f);
  unsigned r = u + 0x7FFFu + ((u >> 16) & 1u);   // round-to-nearest-even
  return (unsigned short)(r >> 16);
}
__device__ __forceinline__ float bf2f(unsigned short h) {
  unsigned u = ((unsigned)h) << 16;
  return __builtin_bit_cast(float, u);
}

// ---- CDNA5 async global->LDS copy (16B per lane), tracked by ASYNCcnt ----
__device__ __forceinline__ void async_copy_b128(unsigned lds_byte_off,
                                                const void* gptr) {
  asm volatile("global_load_async_to_lds_b128 %0, %1, off"
               :
               : "v"(lds_byte_off), "v"(gptr)
               : "memory");
}
__device__ __forceinline__ void wait_asynccnt0() {
  asm volatile("s_wait_asynccnt 0x0" ::: "memory");
}
__device__ __forceinline__ unsigned lds_off32(const void* shared_ptr) {
  // LDS aperture keeps the DS byte offset in the low 32 bits of a flat addr
  return (unsigned)(unsigned long long)shared_ptr;
}

// --------------------------- conversions to bf16 ---------------------------
__global__ void __launch_bounds__(256)
cvt_weights5_kernel(const float* __restrict__ Wq, const float* __restrict__ Wk,
                    const float* __restrict__ Wv, const float* __restrict__ Wo,
                    const float* __restrict__ Wm,
                    unsigned short* __restrict__ wqb, unsigned short* __restrict__ wkb,
                    unsigned short* __restrict__ wvb, unsigned short* __restrict__ wob,
                    unsigned short* __restrict__ wmb) {
  int i = blockIdx.x * 256 + threadIdx.x;
  if (i < DMODEL * DMODEL) {
    wqb[i] = f2bf_u16(Wq[i]);
    wkb[i] = f2bf_u16(Wk[i]);
    wvb[i] = f2bf_u16(Wv[i]);
    wob[i] = f2bf_u16(Wo[i]);
    wmb[i] = f2bf_u16(Wm[i]);
  }
}

__global__ void __launch_bounds__(256)
cvt_memory_kernel(const float* __restrict__ mem, unsigned short* __restrict__ memb) {
  int i = blockIdx.x * 256 + threadIdx.x;
  if (i < N_NODES * DMODEL) memb[i] = f2bf_u16(mem[i]);
}

// ----------------- batched GEMM: C = (A @ W^T + bias) * scale ---------------
// A: [M x 512] bf16 row-major; W: [512 x 512] bf16 row-major; 128 rows/block.
// A tile staged in LDS via async-LDS loads. OUT_BF16 selects output type.
template <bool OUT_BF16>
__global__ void __launch_bounds__(256)
gemm_bf16_kernel(const unsigned short* __restrict__ A,
                 const unsigned short* __restrict__ Wb,
                 const float* __restrict__ bias, float scale,
                 float* __restrict__ outf, unsigned short* __restrict__ outb) {
  extern __shared__ char smem_raw[];
  unsigned short* sA = (unsigned short*)smem_raw;      // 128 * SMS bf16

  const int t = threadIdx.x;
  const int wv = t >> 5, lane = t & 31;
  const int lrow = lane & 15, lhalf = lane >> 4;
  const size_t rowbase = (size_t)blockIdx.x * 128;

  // stage 128 rows x first 256 cols of bf16 A tile into LDS with async loads (16B chunks)
  {
    const unsigned sA_base = lds_off32(sA);
    for (int idx = t; idx < 128 * 32; idx += 256) {
      const int r = idx >> 5, c = idx & 31;            // chunk = 8 bf16
      async_copy_b128(sA_base + (unsigned)(r * SMS + c * 8) * 2u,
                      A + (rowbase + r) * DMODEL + c * 8);
    }
    wait_asynccnt0();
  }
  __syncthreads();

  const __bf16* sAb = (const __bf16*)sA;
  for (int p = 0; p < 4; ++p) {
    const int nt = p * 8 + wv;
    const int ocol = nt * 16 + lrow;
    f32x8 acc[8] = {};
    const __bf16* wrow = (const __bf16*)Wb + (size_t)ocol * DMODEL + lhalf * 16;

    for (int kk = 0; kk < 16; ++kk) {
      bf16x16 bfrag = *(const bf16x16*)(wrow + kk * 32);
      #pragma unroll
      for (int mt = 0; mt < 8; ++mt) {
        const __bf16* ap = sAb + (mt * 16 + lrow) * SMS + kk * 32 + lhalf * 16;
        AFrag a;
        a.h[0] = *(const bf16x8*)ap;
        a.h[1] = *(const bf16x8*)(ap + 8);
        acc[mt] = __builtin_amdgcn_wmma_f32_16x16x32_bf16(
            false, a.v, false, bfrag, (short)0, acc[mt], false, false);
      }
    }

    const float b = bias[ocol];
    #pragma unroll
    for (int mt = 0; mt < 8; ++mt) {
      #pragma unroll
      for (int r = 0; r < 8; ++r) {
        const size_t row = rowbase + mt * 16 + lhalf * 8 + r;
        const float vres = (acc[mt][r] + b) * scale;
        if (OUT_BF16) outb[row * DMODEL + ocol] = f2bf_u16(vres);
        else          outf[row * DMODEL + ocol] = vres;
      }
    }
  }
}

// ------------------- fused per-node projection: dst = msg @ W^T + b --------
__device__ __forceinline__ void proj_gemm_bf16(
    const unsigned short* __restrict__ Wb, const float* __restrict__ biasv,
    const unsigned short* __restrict__ smsg, unsigned short* __restrict__ dst,
    int wv, int lane) {
  const int lrow  = lane & 15;
  const int lhalf = lane >> 4;
  const __bf16* msgb = (const __bf16*)smsg;

  for (int p = 0; p < 4; ++p) {
    const int nt   = p * 8 + wv;
    const int ocol = nt * 16 + lrow;
    f32x8 acc[8] = {};
    const __bf16* wrow = (const __bf16*)Wb + (size_t)ocol * DMODEL + lhalf * 16;

    for (int kk = 0; kk < 16; ++kk) {
      bf16x16 bfrag = *(const bf16x16*)(wrow + kk * 32);
      #pragma unroll
      for (int mt = 0; mt < 8; ++mt) {
        const __bf16* ap = msgb + (mt * 16 + lrow) * SMS + kk * 32 + lhalf * 16;
        AFrag a;
        a.h[0] = *(const bf16x8*)ap;
        a.h[1] = *(const bf16x8*)(ap + 8);
        acc[mt] = __builtin_amdgcn_wmma_f32_16x16x32_bf16(
            false, a.v, false, bfrag, (short)0, acc[mt], false, false);
      }
    }

    const float bias = biasv[ocol];
    #pragma unroll
    for (int mt = 0; mt < 8; ++mt) {
      #pragma unroll
      for (int r = 0; r < 8; ++r) {
        const int row = mt * 16 + lhalf * 8 + r;   // D layout: M = r + 8*(lane/16)
        dst[row * SMS + ocol] = f2bf_u16(acc[mt][r] + bias);
      }
    }
  }
}

// ------------------------- fused attention kernel --------------------------
__global__ void __launch_bounds__(256, 1)
attn_kernel(const float* __restrict__ Qf,
            const float* __restrict__ messages,
            const int*   __restrict__ lengths,
            const unsigned short* __restrict__ Wkb, const float* __restrict__ bk,
            const unsigned short* __restrict__ Wvb, const float* __restrict__ bv,
            unsigned short* __restrict__ ctxb) {
  extern __shared__ char smem_raw[];
  unsigned short* smsg = (unsigned short*)smem_raw;       // 128*SMS bf16
  unsigned short* skv  = smsg + SEQ_L * SMS;              // 128*SMS bf16 (K then V)
  float* sq      = (float*)(skv + SEQ_L * SMS);           // 512
  float* sScores = sq + DMODEL;                           // 1024
  float* sSum    = sScores + NHEAD * SEQ_L;               // 16
  float* sBk     = sSum + 16;                             // 512
  float* sBv     = sBk + DMODEL;                          // 512

  const int n    = blockIdx.x;
  const int t    = threadIdx.x;
  const int wv   = t >> 5;
  const int lane = t & 31;
  const int len  = lengths[n];

  // ---- phase 0a: async-stage q row and biases (pure copies, no convert) ----
  if (t < 128) {
    const float* qrow = Qf + (size_t)n * DMODEL;
    async_copy_b128(lds_off32(sq)  + (unsigned)t * 16u, qrow + t * 4);
    async_copy_b128(lds_off32(sBk) + (unsigned)t * 16u, bk + t * 4);
    async_copy_b128(lds_off32(sBv) + (unsigned)t * 16u, bv + t * 4);
  }

  // ---- phase 0b: bf16-convert this node's messages into LDS ----
  {
    const float* msg = messages + (size_t)n * SEQ_L * DMODEL;
    for (int idx = t; idx < SEQ_L * DMODEL; idx += 256) {
      __builtin_prefetch(msg + idx + 4096, 0, 0);
      const int row = idx >> 9, col = idx & (DMODEL - 1);
      smsg[row * SMS + col] = f2bf_u16(msg[idx]);
    }
  }
  wait_asynccnt0();
  __syncthreads();

  // ---- phase 1: K = msg @ Wk^T + bk (WMMA) ----
  proj_gemm_bf16(Wkb, sBk, smsg, skv, wv, lane);
  __syncthreads();

  // ---- phase 2: scores[h][l] = q_h . k_{l,h}, masked ----
  #pragma unroll
  for (int j = 0; j < 4; ++j) {
    const int idx = t + 256 * j;          // idx = h*128 + l
    const int h = idx >> 7, l = idx & (SEQ_L - 1);
    const float* qh = sq + h * HDIM;
    const unsigned short* krow = skv + l * SMS + h * HDIM;
    float s = 0.f;
    #pragma unroll 8
    for (int d = 0; d < HDIM; ++d) s += qh[d] * bf2f(krow[d]);
    sScores[idx] = (l < len) ? s : -1e30f;
  }
  __syncthreads();

  // ---- phase 3: softmax over l, one wave per head ----
  {
    const int h = wv;
    float v[4];
    #pragma unroll
    for (int j = 0; j < 4; ++j) v[j] = sScores[h * SEQ_L + lane + 32 * j];
    float m = fmaxf(fmaxf(v[0], v[1]), fmaxf(v[2], v[3]));
    #pragma unroll
    for (int off = 16; off >= 1; off >>= 1) m = fmaxf(m, __shfl_xor(m, off, 32));
    float sum = 0.f;
    #pragma unroll
    for (int j = 0; j < 4; ++j) {
      const float p = __expf(v[j] - m);
      sum += p;
      sScores[h * SEQ_L + lane + 32 * j] = p;
    }
    #pragma unroll
    for (int off = 16; off >= 1; off >>= 1) sum += __shfl_xor(sum, off, 32);
    if (lane == 0) sSum[h] = sum;
  }
  __syncthreads();

  // ---- phase 4: V = msg @ Wv^T + bv (WMMA), overwrites K buffer ----
  proj_gemm_bf16(Wvb, sBv, smsg, skv, wv, lane);
  __syncthreads();

  // ---- phase 5: ctx[o] = (1/s_h) * sum_l attn[h][l] * V[l][o] -> bf16 out ----
  #pragma unroll
  for (int j = 0; j < 2; ++j) {
    const int o = t + 256 * j;
    const int h = o >> 6;
    const float* attn = sScores + h * SEQ_L;
    float acc = 0.f;
    #pragma unroll 8
    for (int l = 0; l < SEQ_L; ++l) acc += attn[l] * bf2f(skv[l * SMS + o]);
    ctxb[(size_t)n * DMODEL + o] = f2bf_u16(acc / sSum[h]);
  }
}

// ------------------------------- launcher ----------------------------------
extern "C" void kernel_launch(void* const* d_in, const int* in_sizes, int n_in,
                              void* d_out, int out_size, void* d_ws, size_t ws_size,
                              hipStream_t stream) {
  const float* memory   = (const float*)d_in[0];
  const float* messages = (const float*)d_in[1];
  const int*   lengths  = (const int*)d_in[2];
  const float* Wq = (const float*)d_in[3];
  const float* bq = (const float*)d_in[4];
  const float* Wk = (const float*)d_in[5];
  const float* bk = (const float*)d_in[6];
  const float* Wv = (const float*)d_in[7];
  const float* bv = (const float*)d_in[8];
  const float* Wo = (const float*)d_in[9];
  const float* bo = (const float*)d_in[10];
  const float* Wm = (const float*)d_in[11];
  const float* bm = (const float*)d_in[12];
  float* out = (float*)d_out;

  // workspace layout (ushort elems unless noted)
  const size_t WSZ = (size_t)DMODEL * DMODEL;     // 262144
  const size_t MSZ = (size_t)N_NODES * DMODEL;    // 2097152
  unsigned short* wqb  = (unsigned short*)d_ws;
  unsigned short* wkb  = wqb + WSZ;
  unsigned short* wvb  = wkb + WSZ;
  unsigned short* wob  = wvb + WSZ;
  unsigned short* wmb  = wob + WSZ;
  unsigned short* memb = wmb + WSZ;               // [N, 512] bf16
  unsigned short* ctxb = memb + MSZ;              // [N, 512] bf16
  unsigned short* aggb = ctxb + MSZ;              // [N, 512] bf16
  float*          Qf   = (float*)(aggb + MSZ);    // [N, 512] f32

  cvt_weights5_kernel<<<dim3((DMODEL * DMODEL + 255) / 256), dim3(256), 0, stream>>>(
      Wq, Wk, Wv, Wo, Wm, wqb, wkb, wvb, wob, wmb);
  cvt_memory_kernel<<<dim3((N_NODES * DMODEL + 255) / 256), dim3(256), 0, stream>>>(
      memory, memb);

  const size_t gemm_smem = (size_t)128 * SMS * sizeof(unsigned short);   // 130 KB
  const dim3 gemm_grid(N_NODES / 128);

  // Q = (memory @ Wq^T + bq) * 0.125   (f32 output)
  gemm_bf16_kernel<false><<<gemm_grid, dim3(256), gemm_smem, stream>>>(
      memb, wqb, bq, 0.125f, Qf, nullptr);

  const size_t attn_smem =
      (size_t)2 * SEQ_L * SMS * sizeof(unsigned short) +
      (size_t)(DMODEL + NHEAD * SEQ_L + 16 + 2 * DMODEL) * sizeof(float);

  attn_kernel<<<dim3(N_NODES), dim3(256), attn_smem, stream>>>(
      Qf, messages, lengths, wkb, bk, wvb, bv, ctxb);

  // agg = ctx @ Wo^T + bo  (bf16 output), out = agg @ Wm^T + bm (f32 output)
  gemm_bf16_kernel<true><<<gemm_grid, dim3(256), gemm_smem, stream>>>(
      ctxb, wob, bo, 1.0f, nullptr, aggb);
  gemm_bf16_kernel<false><<<gemm_grid, dim3(256), gemm_smem, stream>>>(
      aggb, wmb, bm, 1.0f, out, nullptr);
}
